// CausalMultiHeadAttention_13692355740471
// MI455X (gfx1250) — matrix operand order repaired and checked by reading, not compiled, not run on hardware
//
#include <hip/hip_runtime.h>

// Causal MHA for MI455X (gfx1250, wave32, WMMA).
// B=4, S=2048, D=1024, H=16, Dh=64. All matmuls via v_wmma_f32_16x16x32_bf16
// (bf16 operands, f32 accumulate); softmax in f32.
// Workspace layout (bf16 bits as ushort): Q[B,H,S,Dh] | K[B,H,S,Dh] | Vt[B,H,Dh,S] | ctx[B,S,D]
// Requires ws_size >= 4 * 8192*1024 * 2 bytes = 64 MiB.

#define B_  4
#define S_  2048
#define D_  1024
#define H_  16
#define DH_ 64
#define M_  (B_ * S_)   // 8192 rows

typedef __attribute__((ext_vector_type(16))) __bf16        v16bf;
typedef __attribute__((ext_vector_type(8)))  float         v8f;
typedef __attribute__((ext_vector_type(4)))  float         f32x4;
typedef __attribute__((ext_vector_type(4)))  unsigned short u16x4;

union Frag {            // 16 bf16 operand fragment = 8 VGPRs, filled by two b128 loads
  v16bf v;
  f32x4 f4[2];
};

__device__ __forceinline__ unsigned short f2bf_u(float x) {
  union { float f; unsigned u; } a; a.f = x;
  unsigned u = a.u;
  u += 0x7FFFu + ((u >> 16) & 1u);            // round-to-nearest-even
  return (unsigned short)(u >> 16);
}

// ---------------------------------------------------------------------------
// Tiled GEMM: out = A[M,1024] @ W[1024,1024] + bias, 128x64 block tile,
// 8 waves/block, each wave owns a 16x64 strip (4 accumulators).
// MODE 0: A=f32 x, out=bf16 [B,H,S,Dh]   (Q and K projections)
// MODE 1: A=f32 x, out=bf16 [B,H,Dh,S]   (V projection, transposed)
// MODE 2: A=bf16 ctx, out=f32 [M,1024]   (output projection)
// ---------------------------------------------------------------------------
template <int MODE>
__global__ __launch_bounds__(256) void gemm_wmma(
    const float* __restrict__ Af, const unsigned short* __restrict__ Ab,
    const float* __restrict__ W, const float* __restrict__ bias,
    unsigned short* __restrict__ outb, float* __restrict__ outf)
{
  __shared__ unsigned short lA[128 * 32];   // [row][k]  8 KB
  __shared__ unsigned short lB[64 * 32];    // [n][k]    4 KB

  const int t    = threadIdx.x;
  const int lane = t & 31;
  const int wv   = t >> 5;      // wave 0..7
  const int g    = lane >> 4;   // half-wave group
  const int ln   = lane & 15;

  const int m0 = (blockIdx.x & 63) * 128;   // M_/128 = 64 block rows
  const int n0 = (blockIdx.x >> 6) * 64;    // D_/64  = 16 block cols

  const v8f vzero = {0.f,0.f,0.f,0.f,0.f,0.f,0.f,0.f};
  v8f acc[4];
  #pragma unroll
  for (int i = 0; i < 4; ++i) acc[i] = vzero;

  for (int k0 = 0; k0 < D_; k0 += 32) {
    // ---- stage A tile (128x32) to LDS as bf16, row-major
    #pragma unroll
    for (int i = 0; i < 4; ++i) {
      int gi  = t + i * 256;          // 1024 groups of 4 elements
      int row = gi >> 3;
      int kc  = (gi & 7) << 2;
      if (MODE == 2) {
        u16x4 d = *(const u16x4*)(Ab + (size_t)(m0 + row) * D_ + k0 + kc);
        *(u16x4*)&lA[row * 32 + kc] = d;
      } else {
        f32x4 s4 = *(const f32x4*)(Af + (size_t)(m0 + row) * D_ + k0 + kc);
        u16x4 d;
        d[0] = f2bf_u(s4[0]); d[1] = f2bf_u(s4[1]);
        d[2] = f2bf_u(s4[2]); d[3] = f2bf_u(s4[3]);
        *(u16x4*)&lA[row * 32 + kc] = d;
      }
    }
    // ---- stage B tile (32k x 64n) to LDS transposed -> [n][k]
    #pragma unroll
    for (int i = 0; i < 8; ++i) {
      int gi = t + i * 256;           // 2048 elements
      int n  = gi & 63;
      int k  = gi >> 6;
      lB[n * 32 + k] = f2bf_u(W[(size_t)(k0 + k) * D_ + n0 + n]);
    }
    __syncthreads();

    // ---- A fragment: lane row = wv*16+ln, K chunks {g*8..+7, 16+g*8..+7}
    const unsigned short* aRow = &lA[(wv * 16 + ln) * 32];
    Frag fa;
    fa.f4[0] = *(const f32x4*)(aRow + g * 8);
    fa.f4[1] = *(const f32x4*)(aRow + 16 + g * 8);

    #pragma unroll
    for (int nt = 0; nt < 4; ++nt) {
      // B fragment: lane column = nt*16+ln, K chunks {g*8..+7, 16+g*8..+7}
      const unsigned short* bCol = &lB[(nt * 16 + ln) * 32];
      Frag fb;
      fb.f4[0] = *(const f32x4*)(bCol + g * 8);
      fb.f4[1] = *(const f32x4*)(bCol + 16 + g * 8);
      acc[nt] = __builtin_amdgcn_wmma_f32_16x16x32_bf16(
          false, fa.v, false, fb.v, (short)0, acc[nt], false, false);
    }
    __syncthreads();
  }

  // ---- epilogue: bias add + store per layout
  #pragma unroll
  for (int nt = 0; nt < 4; ++nt) {
    int col = n0 + nt * 16 + ln;
    float bvv = bias[col];
    #pragma unroll
    for (int r = 0; r < 8; ++r) {
      int m   = m0 + wv * 16 + g * 8 + r;     // C row = r + 8*g
      float val = acc[nt][r] + bvv;
      if (MODE == 2) {
        outf[(size_t)m * D_ + col] = val;
      } else {
        int b = m >> 11, s = m & (S_ - 1);
        int h = col >> 6, d = col & 63;
        size_t off = (MODE == 0)
            ? ((size_t)(b * H_ + h) * S_ + s) * DH_ + d    // [B,H,S,Dh]
            : ((size_t)(b * H_ + h) * DH_ + d) * S_ + s;   // [B,H,Dh,S]
        outb[off] = f2bf_u(val);
      }
    }
  }
}

// ---------------------------------------------------------------------------
// Flash-style causal attention: one wave per (b,h, 16-query tile).
// Streams 32-key tiles: scores = Q Kt (4 WMMA), online softmax with
// half-wave shfl row reductions, P through LDS (C-layout -> A-layout),
// ctx += P V (4 WMMA).
// ---------------------------------------------------------------------------
__global__ __launch_bounds__(32) void attn_wmma(
    const unsigned short* __restrict__ Q,
    const unsigned short* __restrict__ Kv,
    const unsigned short* __restrict__ Vt,
    unsigned short* __restrict__ Ctx)
{
  __shared__ unsigned short lP[16 * 32];     // P tile [16 q][32 k] bf16, 1 KB

  const int lane = threadIdx.x;
  const int g    = lane >> 4;
  const int ln   = lane & 15;
  const int qt   = blockIdx.x & (S_ / 16 - 1);   // 128 q-tiles
  const int bh   = blockIdx.x >> 7;
  const int q0   = qt * 16;

  // Q A-fragments for both 32-wide d-chunks (kept live all loop)
  const unsigned short* qrow = Q + ((size_t)bh * S_ + q0 + ln) * DH_;
  Frag qa[2];
  #pragma unroll
  for (int c = 0; c < 2; ++c) {
    qa[c].f4[0] = *(const f32x4*)(qrow + c * 32 + g * 8);
    qa[c].f4[1] = *(const f32x4*)(qrow + c * 32 + 16 + g * 8);
  }

  const v8f vzero = {0.f,0.f,0.f,0.f,0.f,0.f,0.f,0.f};
  v8f acc[4];
  #pragma unroll
  for (int i = 0; i < 4; ++i) acc[i] = vzero;
  float rm[8], rs[8];
  #pragma unroll
  for (int r = 0; r < 8; ++r) { rm[r] = -3.0e38f; rs[r] = 0.0f; }

  const int nk = (q0 + 16 + 31) >> 5;        // only keys <= last query
  for (int kt = 0; kt < nk; ++kt) {
    const int sk0 = kt * 32;

    // ---- scores: two 16x16 tiles, K-dim = Dh = 64 (two bf16 WMMAs each)
    v8f sc[2] = {vzero, vzero};
    #pragma unroll
    for (int tt = 0; tt < 2; ++tt) {
      const unsigned short* krow = Kv + ((size_t)bh * S_ + sk0 + tt * 16 + ln) * DH_;
      #pragma unroll
      for (int c = 0; c < 2; ++c) {
        Frag kb;   // B-frag: column = key, K = d = c*32 + {g*8..+7, 16+g*8..+7}
        kb.f4[0] = *(const f32x4*)(krow + c * 32 + g * 8);
        kb.f4[1] = *(const f32x4*)(krow + c * 32 + 16 + g * 8);
        sc[tt] = __builtin_amdgcn_wmma_f32_16x16x32_bf16(
            false, qa[c].v, false, kb.v, (short)0, sc[tt], false, false);
      }
    }

    // ---- causal mask + online softmax (rows live across 16 lanes)
    #pragma unroll
    for (int r = 0; r < 8; ++r) {
      const int q = q0 + g * 8 + r;
      float v0[2];
      float tmax = -3.0e38f;
      #pragma unroll
      for (int tt = 0; tt < 2; ++tt) {
        int key = sk0 + tt * 16 + ln;
        float v = sc[tt][r] * 0.125f;               // 1/sqrt(64)
        v0[tt] = (key > q) ? -3.0e38f : v;
        tmax = fmaxf(tmax, v0[tt]);
      }
      #pragma unroll
      for (int mk = 1; mk < 16; mk <<= 1)
        tmax = fmaxf(tmax, __shfl_xor(tmax, mk, 32));
      float newm  = fmaxf(rm[r], tmax);
      float scale = __expf(rm[r] - newm);
      rm[r] = newm;
      float p0 = __expf(v0[0] - newm);
      float p1 = __expf(v0[1] - newm);
      float ps = p0 + p1;
      #pragma unroll
      for (int mk = 1; mk < 16; mk <<= 1)
        ps += __shfl_xor(ps, mk, 32);
      rs[r] = rs[r] * scale + ps;
      #pragma unroll
      for (int nt = 0; nt < 4; ++nt) acc[nt][r] *= scale;
      const int mrow = g * 8 + r;
      lP[mrow * 32 + ln]      = f2bf_u(p0);
      lP[mrow * 32 + 16 + ln] = f2bf_u(p1);
    }
    __syncthreads();   // single-wave WG: orders LDS stores before reads

    // ---- ctx += P[16x32] @ V[32x64]
    const unsigned short* prow = lP + ln * 32;
    Frag pa;
    pa.f4[0] = *(const f32x4*)(prow + g * 8);
    pa.f4[1] = *(const f32x4*)(prow + 16 + g * 8);
    #pragma unroll
    for (int nt = 0; nt < 4; ++nt) {
      // B-frag from V^T: column = d, K = key = sk0 + {g*8..+7, 16+g*8..+7}
      const unsigned short* vcol =
          Vt + ((size_t)bh * DH_ + nt * 16 + ln) * S_ + sk0;
      Frag vb;
      vb.f4[0] = *(const f32x4*)(vcol + g * 8);
      vb.f4[1] = *(const f32x4*)(vcol + 16 + g * 8);
      acc[nt] = __builtin_amdgcn_wmma_f32_16x16x32_bf16(
          false, pa.v, false, vb.v, (short)0, acc[nt], false, false);
    }
    __syncthreads();   // protect lP before next iteration's stores
  }

  // ---- normalize and store ctx as bf16 [B,S,D] (ready as A for out-proj)
  const int b = bh >> 4, h = bh & 15;
  #pragma unroll
  for (int r = 0; r < 8; ++r) {
    const float inv = 1.0f / rs[r];
    const int q = q0 + g * 8 + r;
    unsigned short* crow = Ctx + ((size_t)b * S_ + q) * D_ + h * DH_;
    #pragma unroll
    for (int nt = 0; nt < 4; ++nt)
      crow[nt * 16 + ln] = f2bf_u(acc[nt][r] * inv);
  }
}

// ---------------------------------------------------------------------------
extern "C" void kernel_launch(void* const* d_in, const int* in_sizes, int n_in,
                              void* d_out, int out_size, void* d_ws, size_t ws_size,
                              hipStream_t stream) {
  const float* x  = (const float*)d_in[0];
  const float* wq = (const float*)d_in[1];
  const float* bq = (const float*)d_in[2];
  const float* wk = (const float*)d_in[3];
  const float* bk = (const float*)d_in[4];
  const float* wv = (const float*)d_in[5];
  const float* bv = (const float*)d_in[6];
  const float* wo = (const float*)d_in[7];
  const float* bo = (const float*)d_in[8];
  float* out = (float*)d_out;

  unsigned short* ws = (unsigned short*)d_ws;
  const size_t SZ = (size_t)M_ * D_;          // 8,388,608 elements each
  unsigned short* Qb = ws;
  unsigned short* Kb = ws + SZ;
  unsigned short* Vb = ws + 2 * SZ;
  unsigned short* Cb = ws + 3 * SZ;           // total 64 MiB of d_ws

  const dim3 gblk(256);
  const dim3 ggrd((M_ / 128) * (D_ / 64));    // 1024 blocks

  gemm_wmma<0><<<ggrd, gblk, 0, stream>>>(x, nullptr, wq, bq, Qb, nullptr);
  gemm_wmma<0><<<ggrd, gblk, 0, stream>>>(x, nullptr, wk, bk, Kb, nullptr);
  gemm_wmma<1><<<ggrd, gblk, 0, stream>>>(x, nullptr, wv, bv, Vb, nullptr);

  attn_wmma<<<dim3(B_ * H_ * (S_ / 16)), dim3(32), 0, stream>>>(Qb, Kb, Vb, Cb);

  gemm_wmma<2><<<ggrd, gblk, 0, stream>>>(nullptr, Cb, wo, bo, nullptr, out);
}
